// final_model_RGAT_80668075754165
// MI455X (gfx1250) — matrix operand order repaired and checked by reading, not compiled, not run on hardware
//
#include <hip/hip_runtime.h>
#include <math.h>

// ---------------------------------------------------------------------------
// RGAT pipeline for MI455X (gfx1250, wave32).
//  - Dense GEMMs: v_wmma_f32_16x16x32_f16 (fp16 in, fp32 accumulate).
//  - Graph is fixed per call -> build dst-CSR once per relation (u32 atomics +
//    one-workgroup LDS scan), reused by all 6 RGAT invocations.
//  - Edge softmax + aggregation: owner-computes, one wave per (node, head);
//    wave reductions via __shfl_xor, per-edge alpha broadcast via __shfl
//    (ds_bpermute), coalesced gathers of h[src]; NO float atomics, single
//    non-atomic write of each accumulator row (~10x less scatter traffic).
//  - adj (144 MB stream-once output) uses nontemporal loads/stores.
// ---------------------------------------------------------------------------

#define N_NODES 20000
#define N1_     6000
#define N2_     6000
#define R_REL   3
#define E_EDGE  200000
#define HID_    128
#define NDIM_   64
#define D1_     64
#define D2_     32
#define H1_     4
#define H2_     2
#define S_      2

typedef __attribute__((ext_vector_type(16))) _Float16 v16h;
typedef __attribute__((ext_vector_type(8)))  float    v8f;

__device__ __forceinline__ float leaky02(float v) { return v > 0.f ? v : 0.2f * v; }

// ---- fills / copies --------------------------------------------------------
__global__ void fill_f32_kernel(float* p, float v, long long n) {
    long long i = (long long)blockIdx.x * blockDim.x + threadIdx.x;
    if (i < n) p[i] = v;
}
__global__ void fill_u32_kernel(unsigned* p, unsigned v, long long n) {
    long long i = (long long)blockIdx.x * blockDim.x + threadIdx.x;
    if (i < n) p[i] = v;
}
__global__ void copy_u32_kernel(const unsigned* __restrict__ a, unsigned* __restrict__ b,
                                long long n) {
    long long i = (long long)blockIdx.x * blockDim.x + threadIdx.x;
    if (i < n) b[i] = a[i];
}

// ---- CSR build (once per relation, reused by all 6 RGAT calls) -------------
__global__ void csr_count_kernel(const int* __restrict__ dst, unsigned* __restrict__ deg,
                                 int E) {
    int e = blockIdx.x * blockDim.x + threadIdx.x;
    if (e < E) atomicAdd(&deg[dst[e]], 1u);
}

#define SCAN_T 1024
// Exclusive prefix sum of deg[0..n) -> rowptr[0..n]; single workgroup.
__global__ void scan_rowptr_kernel(const unsigned* __restrict__ deg,
                                   unsigned* __restrict__ rowptr, int n) {
    __shared__ unsigned part[SCAN_T];
    int t = threadIdx.x;
    int CH = (n + SCAN_T - 1) / SCAN_T;
    int beg = t * CH, end = (beg + CH < n) ? beg + CH : n;
    unsigned s = 0;
    for (int i = beg; i < end; ++i) s += deg[i];
    part[t] = s;
    __syncthreads();
    for (int off = 1; off < SCAN_T; off <<= 1) {     // Hillis-Steele inclusive
        unsigned o = (t >= off) ? part[t - off] : 0u;
        __syncthreads();
        part[t] += o;
        __syncthreads();
    }
    unsigned run = part[t] - s;                      // exclusive chunk base
    for (int i = beg; i < end; ++i) { rowptr[i] = run; run += deg[i]; }
    if (t == SCAN_T - 1) rowptr[n] = run;            // total (= E)
}

// Allocate slots and record the SOURCE node per incoming edge.
__global__ void csr_fill_kernel(const int* __restrict__ src, const int* __restrict__ dst,
                                unsigned* __restrict__ pos, int* __restrict__ srcs,
                                int E) {
    int e = blockIdx.x * blockDim.x + threadIdx.x;
    if (e >= E) return;
    unsigned idx = atomicAdd(&pos[dst[e]], 1u);
    srcs[idx] = src[e];
}

// ---- WMMA GEMM: C[M,N] = A[M,K] (row major f32) @ B[K,N] (row major f32) ----
__global__ void wmma_gemm_kernel(const float* __restrict__ A,
                                 const float* __restrict__ B,
                                 float* __restrict__ C,
                                 int M, int Nn, int K) {
    long long wave = ((long long)blockIdx.x * blockDim.x + threadIdx.x) >> 5;
    int lane = threadIdx.x & 31;
    int mtiles = M >> 4, ntiles = Nn >> 4;
    if (wave >= (long long)mtiles * ntiles) return;
    int mt = (int)(wave / ntiles), nt = (int)(wave % ntiles);
    int row0 = mt << 4, col0 = nt << 4;
    int half = lane >> 4, lrow = lane & 15;

    v8f c = {};
    for (int k0 = 0; k0 < K; k0 += 32) {
        // A fragment: lane holds row (row0+lrow), K = half*16 + {0..15}
        const float* ap = A + (size_t)(row0 + lrow) * K + k0 + half * 16;
        v16h a, b;
#pragma unroll
        for (int j = 0; j < 8; ++j) a[j]     = (_Float16)ap[j];
#pragma unroll
        for (int j = 0; j < 8; ++j) a[8 + j] = (_Float16)ap[8 + j];
        // B fragment: lane holds col (col0+lrow), K = half*16 + j
        const float* bp = B + (size_t)(k0 + half * 16) * Nn + col0 + lrow;
#pragma unroll
        for (int j = 0; j < 16; ++j) b[j] = (_Float16)bp[(size_t)j * Nn];
        c = __builtin_amdgcn_wmma_f32_16x16x32_f16(false, a, false, b,
                                                   (short)0, c, false, false);
    }
    float* cp = C + (size_t)(row0 + half * 8) * Nn + col0 + lrow;
#pragma unroll
    for (int i = 0; i < 8; ++i) cp[(size_t)i * Nn] = c[i];
}

// ---- adj GEMM: out = (accum? out:0) + scale*sigmoid(z1[M,K] @ z2[N,K]^T) ---
__global__ void wmma_adj_kernel(const float* __restrict__ z1,
                                const float* __restrict__ z2,
                                float* __restrict__ out,
                                int M, int Nn, int K, float scale, int accum) {
    long long wave = ((long long)blockIdx.x * blockDim.x + threadIdx.x) >> 5;
    int lane = threadIdx.x & 31;
    int mtiles = M >> 4, ntiles = Nn >> 4;
    if (wave >= (long long)mtiles * ntiles) return;
    int mt = (int)(wave / ntiles), nt = (int)(wave % ntiles);
    int row0 = mt << 4, col0 = nt << 4;
    int half = lane >> 4, lrow = lane & 15;

    v8f c = {};
    for (int k0 = 0; k0 < K; k0 += 32) {
        const float* ap = z1 + (size_t)(row0 + lrow) * K + k0 + half * 16;
        v16h a, b;
#pragma unroll
        for (int j = 0; j < 8; ++j) a[j]     = (_Float16)ap[j];
#pragma unroll
        for (int j = 0; j < 8; ++j) a[8 + j] = (_Float16)ap[8 + j];
        const float* bp = z2 + (size_t)(col0 + lrow) * K + k0 + half * 16;
#pragma unroll
        for (int j = 0; j < 16; ++j) b[j] = (_Float16)bp[j];
        c = __builtin_amdgcn_wmma_f32_16x16x32_f16(false, a, false, b,
                                                   (short)0, c, false, false);
    }
    float* cp = out + (size_t)(row0 + half * 8) * Nn + col0 + lrow;
#pragma unroll
    for (int i = 0; i < 8; ++i) {
        float v = scale / (1.0f + __expf(-c[i]));
        float* p = cp + (size_t)i * Nn;
        if (accum) v += __builtin_nontemporal_load(p);
        __builtin_nontemporal_store(v, p);
    }
}

// ---- attention pieces ------------------------------------------------------
template <int H, int F>
__global__ void eler_kernel(const float* __restrict__ h,
                            const float* __restrict__ al,
                            const float* __restrict__ ar,
                            float* __restrict__ el, float* __restrict__ er,
                            int n) {
    int idx = blockIdx.x * blockDim.x + threadIdx.x;  // over n*H
    if (idx >= n * H) return;
    int node = idx / H, hh = idx % H;
    const float* hp  = h  + (size_t)node * H * F + (size_t)hh * F;
    const float* alp = al + (size_t)hh * F;
    const float* arp = ar + (size_t)hh * F;
    float sl = 0.f, sr = 0.f;
#pragma unroll 8
    for (int f = 0; f < F; ++f) { float v = hp[f]; sl += v * alp[f]; sr += v * arp[f]; }
    el[idx] = sl; er[idx] = sr;
}

// Owner-computes edge softmax + aggregation: one wave per (node, head).
// 3 in-register passes over the node's incoming edges (max, denom, accum).
// acc row is updated exactly once, non-atomically (accumulates over relations).
template <int H, int F>
__global__ void node_attn_kernel(const unsigned* __restrict__ rowptr,
                                 const int* __restrict__ srcs,
                                 const float* __restrict__ el,
                                 const float* __restrict__ er,
                                 const float* __restrict__ hbuf,
                                 float* __restrict__ acc, int n) {
    constexpr int HF  = H * F;
    constexpr int PER = F / 32;                 // 2 for F=64, 1 for F=32
    long long wave = ((long long)blockIdx.x * blockDim.x + threadIdx.x) >> 5;
    int lane = threadIdx.x & 31;
    if (wave >= (long long)n * H) return;
    int node = (int)(wave / H), hh = (int)(wave % H);

    int beg = (int)rowptr[node], end = (int)rowptr[node + 1];
    float er_d = er[node * H + hh];

    // pass 1: segment max
    float m = -3.4e38f;
    for (int base = beg; base < end; base += 32) {
        int i = base + lane;
        float v = -3.4e38f;
        if (i < end) v = leaky02(el[srcs[i] * H + hh] + er_d);
        m = fmaxf(m, v);
    }
#pragma unroll
    for (int off = 16; off; off >>= 1) m = fmaxf(m, __shfl_xor(m, off));
    if (m < -3.0e38f) m = 0.f;                  // where(isfinite(m), m, 0)

    // pass 2: denominator
    float dsum = 0.f;
    for (int base = beg; base < end; base += 32) {
        int i = base + lane;
        if (i < end) dsum += __expf(leaky02(el[srcs[i] * H + hh] + er_d) - m);
    }
#pragma unroll
    for (int off = 16; off; off >>= 1) dsum += __shfl_xor(dsum, off);
    float inv = 1.0f / (dsum + 1e-16f);

    // pass 3: weighted feature accumulation
    float facc[PER];
#pragma unroll
    for (int q = 0; q < PER; ++q) facc[q] = 0.f;

    for (int base = beg; base < end; base += 32) {
        int i = base + lane;
        int   s = 0;
        float w = 0.f;
        if (i < end) {
            s = srcs[i];
            w = __expf(leaky02(el[s * H + hh] + er_d) - m);
        }
        int cnt = end - base; if (cnt > 32) cnt = 32;
        for (int j = 0; j < cnt; ++j) {         // wave-uniform loop
            float alpha = __shfl(w, j) * inv;
            int   sj    = __shfl(s, j);
            const float* hrow = hbuf + (size_t)sj * HF + hh * F;
#pragma unroll
            for (int q = 0; q < PER; ++q)
                facc[q] += alpha * hrow[lane + q * 32];
        }
    }

    float* ap = acc + (size_t)node * HF + hh * F;
#pragma unroll
    for (int q = 0; q < PER; ++q) ap[lane + q * 32] += facc[q];
}

// dst[n,f] = act( mean_h acc[n,h,f] ) (+ addend[n,f])
template <int H, int F>
__global__ void finalize_kernel(const float* __restrict__ acc,
                                const float* __restrict__ addend,
                                float* __restrict__ dst,
                                int n, int do_relu) {
    int idx = blockIdx.x * blockDim.x + threadIdx.x;  // over n*F
    if (idx >= n * F) return;
    int node = idx / F, f = idx % F;
    float s = 0.f;
#pragma unroll
    for (int h = 0; h < H; ++h) s += acc[(size_t)node * H * F + (size_t)h * F + f];
    s *= (1.0f / (float)H);
    if (do_relu) s = fmaxf(s, 0.f);
    if (addend)  s += addend[idx];
    dst[idx] = s;
}

__global__ void sigmoid_kernel(const float* __restrict__ in, float* __restrict__ out, int n) {
    int i = blockIdx.x * blockDim.x + threadIdx.x;
    if (i < n) out[i] = 1.0f / (1.0f + __expf(-in[i]));
}

// ---------------------------------------------------------------------------
// Host orchestration
// ---------------------------------------------------------------------------
static inline unsigned blocksFor(long long threads, int bs) {
    return (unsigned)((threads + bs - 1) / bs);
}

// workspace layout (4-byte units)
#define WS_H    ((size_t)0)
#define WS_ACC  (WS_H    + (size_t)N_NODES * H1_ * D1_)     //  5,120,000
#define WS_EL   (WS_ACC  + (size_t)N_NODES * H1_ * D1_)     // 10,240,000
#define WS_ER   (WS_EL   + (size_t)N_NODES * H1_)
#define WS_HX   (WS_ER   + (size_t)N_NODES * H1_)
#define WS_H1S  (WS_HX   + (size_t)N_NODES * D1_)
#define WS_ROW  (WS_H1S  + (size_t)S_ * N_NODES * D1_)      // 3*(N+1) u32
#define WS_SRC  (WS_ROW  + (size_t)R_REL * (N_NODES + 1))   // 3*E u32
#define WS_POS  (WS_SRC  + (size_t)R_REL * E_EDGE)          // N u32 (scratch)
// total ~ 14.9M * 4B ~ 60 MB

template <int H, int F>
static void run_rgat(const float* X, int Din,
                     const float* W, const float* al, const float* ar,
                     const unsigned* rowptrAll, const int* srcsAll,
                     float* ws, float* result, const float* addend, int do_relu,
                     hipStream_t st) {
    float* hbuf = ws + WS_H;
    float* acc  = ws + WS_ACC;
    float* el   = ws + WS_EL;
    float* er   = ws + WS_ER;

    const long long nHF = (long long)N_NODES * H * F;
    const long long nH  = (long long)N_NODES * H;

    fill_f32_kernel<<<blocksFor(nHF, 256), 256, 0, st>>>(acc, 0.f, nHF);

    for (int r = 0; r < R_REL; ++r) {
        const float*    Wr   = W  + (size_t)r * Din * H * F;
        const float*    alr  = al + (size_t)r * H * F;
        const float*    arr  = ar + (size_t)r * H * F;
        const unsigned* rptr = rowptrAll + (size_t)r * (N_NODES + 1);
        const int*      srcs = srcsAll   + (size_t)r * E_EDGE;

        long long waves = (long long)(N_NODES / 16) * ((H * F) / 16);
        wmma_gemm_kernel<<<blocksFor(waves * 32, 256), 256, 0, st>>>(
            X, Wr, hbuf, N_NODES, H * F, Din);

        eler_kernel<H, F><<<blocksFor(nH, 256), 256, 0, st>>>(hbuf, alr, arr,
                                                              el, er, N_NODES);
        node_attn_kernel<H, F><<<blocksFor(nH * 32, 256), 256, 0, st>>>(
            rptr, srcs, el, er, hbuf, acc, N_NODES);
    }

    finalize_kernel<H, F><<<blocksFor((long long)N_NODES * F, 256), 256, 0, st>>>(
        acc, addend, result, N_NODES, do_relu);
}

extern "C" void kernel_launch(void* const* d_in, const int* in_sizes, int n_in,
                              void* d_out, int out_size, void* d_ws, size_t ws_size,
                              hipStream_t stream) {
    (void)in_sizes; (void)n_in; (void)out_size; (void)ws_size;

    const float* x     = (const float*)d_in[0];
    const float* noise = (const float*)d_in[1];
    const float* W1    = (const float*)d_in[2];
    const float* al1   = (const float*)d_in[3];
    const float* ar1   = (const float*)d_in[4];
    const float* We    = (const float*)d_in[5];
    const float* ale   = (const float*)d_in[6];
    const float* are   = (const float*)d_in[7];
    const float* W2    = (const float*)d_in[8];
    const float* al2   = (const float*)d_in[9];
    const float* ar2   = (const float*)d_in[10];
    const float* W3    = (const float*)d_in[11];
    const float* al3   = (const float*)d_in[12];
    const float* ar3   = (const float*)d_in[13];
    const float* rk    = (const float*)d_in[14];
    const int*   src   = (const int*)d_in[15];
    const int*   dstE  = (const int*)d_in[16];

    float* out = (float*)d_out;
    float* ws  = (float*)d_ws;

    // output layout: adj [1,6000,6000] | mu [2,20000,32] | logvar [20000,32] | rk2 [1,32]
    const size_t ADJ_OFF = 0;
    const size_t MU_OFF  = (size_t)N1_ * N2_;
    const size_t LV_OFF  = MU_OFF + (size_t)S_ * N_NODES * D2_;
    const size_t RK_OFF  = LV_OFF + (size_t)N_NODES * D2_;

    float*    hx     = ws + WS_HX;
    float*    h1     = ws + WS_H1S;
    unsigned* rowptr = (unsigned*)(ws + WS_ROW);
    int*      srcs   = (int*)(ws + WS_SRC);
    unsigned* pos    = (unsigned*)(ws + WS_POS);   // doubles as deg scratch

    // ---- build dst-CSR per relation (reused by all 6 RGAT calls) ----
    for (int r = 0; r < R_REL; ++r) {
        unsigned* rptr = rowptr + (size_t)r * (N_NODES + 1);
        int*      srcr = srcs   + (size_t)r * E_EDGE;
        const int* sr  = src  + (size_t)r * E_EDGE;
        const int* dr  = dstE + (size_t)r * E_EDGE;

        fill_u32_kernel<<<blocksFor(N_NODES, 256), 256, 0, stream>>>(pos, 0u, N_NODES);
        csr_count_kernel<<<blocksFor(E_EDGE, 256), 256, 0, stream>>>(dr, pos, E_EDGE);
        scan_rowptr_kernel<<<1, SCAN_T, 0, stream>>>(pos, rptr, N_NODES);
        copy_u32_kernel<<<blocksFor(N_NODES, 256), 256, 0, stream>>>(rptr, pos, N_NODES);
        csr_fill_kernel<<<blocksFor(E_EDGE, 256), 256, 0, stream>>>(sr, dr, pos, srcr,
                                                                    E_EDGE);
    }

    // hiddenx = rgat(x; W1) with relu
    run_rgat<H1_, D1_>(x, HID_, W1, al1, ar1, rowptr, srcs, ws, hx, nullptr, 1, stream);

    // hidden1[s] = hiddenx + relu-rgat(noise[s]; We)
    for (int s = 0; s < S_; ++s) {
        float* h1s = h1 + (size_t)s * N_NODES * D1_;
        run_rgat<H1_, D1_>(noise + (size_t)s * N_NODES * NDIM_, NDIM_, We, ale, are,
                           rowptr, srcs, ws, h1s, hx, 1, stream);
    }

    // mu[s] = rgat(hidden1[s]; W2), identity act, written straight to d_out
    for (int s = 0; s < S_; ++s) {
        float* mus = out + MU_OFF + (size_t)s * N_NODES * D2_;
        run_rgat<H2_, D2_>(h1 + (size_t)s * N_NODES * D1_, D1_, W2, al2, ar2,
                           rowptr, srcs, ws, mus, nullptr, 0, stream);
    }

    // logvar = rgat(hiddenx; W3), identity act
    run_rgat<H2_, D2_>(hx, D1_, W3, al3, ar3, rowptr, srcs, ws,
                       out + LV_OFF, nullptr, 0, stream);

    // rk2 = sigmoid(rk_lgt)
    sigmoid_kernel<<<1, 32, 0, stream>>>(rk, out + RK_OFF, D2_);

    // adj = mean_s sigmoid(z1_s @ z2_s^T), z from mu already in d_out
    for (int s = 0; s < S_; ++s) {
        const float* z1 = out + MU_OFF + (size_t)s * N_NODES * D2_;
        const float* z2 = z1 + (size_t)N1_ * D2_;
        long long tiles = (long long)(N1_ / 16) * (N2_ / 16);
        wmma_adj_kernel<<<blocksFor(tiles * 32, 256), 256, 0, stream>>>(
            z1, z2, out + ADJ_OFF, N1_, N2_, D2_, 0.5f, s);
    }
}
